// BiMPM_20126216749351
// MI455X (gfx1250) — matrix operand order repaired and checked by reading, not compiled, not run on hardware
//
#include <hip/hip_runtime.h>
#include <math.h>

#define EPSV 1e-8f
#define BB   4
#define SQ   256
#define DD   300
#define Hh   256
#define G4   1024
#define LL   10
#define NCLSV 22
#define AGGIN 62

typedef __attribute__((ext_vector_type(16))) _Float16 v16h;
typedef __attribute__((ext_vector_type(8)))  _Float16 v8h;
typedef __attribute__((ext_vector_type(8)))  float    v8f;

__device__ inline float sigm(float x) { return 1.f / (1.f + expf(-x)); }

__device__ inline v16h cat8(v8h lo, v8h hi) {
  v16h r;
#pragma unroll
  for (int i = 0; i < 8; ++i) { r[i] = lo[i]; r[8 + i] = hi[i]; }
  return r;
}

// ---------------------------------------------------------------------------
// Fragment builders (this file's contiguous k order, shared by A and B).
// A (16x32 f16): lane(half, ml) holds row m=ml, one contiguous K-run:
//   k = kk + 16*half + e       (e = 0..15)
// B (32x16 f16): lane(half, ml) holds col n=ml, one contiguous K-run:
//   k = kk + 16*half + e       (e = 0..15)
// ---------------------------------------------------------------------------
__device__ inline v16h frag_a_rowmajor(const float* __restrict__ row, int kk, int half) {
  const float2* q0 = (const float2*)(row + kk + 16 * half);
  const float2* q1 = (const float2*)(row + kk + 16 * half + 8);
  v16h a;
#pragma unroll
  for (int e = 0; e < 4; ++e) {
    float2 x = q0[e]; a[2 * e] = (_Float16)x.x; a[2 * e + 1] = (_Float16)x.y;
    float2 y = q1[e]; a[8 + 2 * e] = (_Float16)y.x; a[8 + 2 * e + 1] = (_Float16)y.y;
  }
  return a;
}

__device__ inline v16h frag_a_kstrided(const float* __restrict__ A, int lda, int gm,
                                       int kk, int half) {
  v16h a;
#pragma unroll
  for (int e = 0; e < 8; ++e) {
    int k0 = kk + 16 * half + e;
    int k1 = kk + 16 * half + 8 + e;
    a[e]     = (_Float16)A[(long long)k0 * lda + gm];
    a[8 + e] = (_Float16)A[(long long)k1 * lda + gm];
  }
  return a;
}

__device__ inline v16h frag_b_rowmajor(const float* __restrict__ row, int kk, int half) {
  const float2* q = (const float2*)(row + kk + 16 * half);
  v16h b;
#pragma unroll
  for (int e = 0; e < 8; ++e) {
    float2 x = q[e]; b[2 * e] = (_Float16)x.x; b[2 * e + 1] = (_Float16)x.y;
  }
  return b;
}

__device__ inline v16h frag_b_kstrided(const float* __restrict__ B, int ldb, int gn,
                                       int kk, int half) {
  v16h b;
#pragma unroll
  for (int e = 0; e < 16; ++e)
    b[e] = (_Float16)B[(long long)(kk + 16 * half + e) * ldb + gn];
  return b;
}

template <int TA>
__device__ inline v16h frag_a_tail(const float* __restrict__ A, int lda, int gm,
                                   int kk, int half, int K) {
  v16h a;
#pragma unroll
  for (int e = 0; e < 16; ++e) {
    int k = kk + 16 * half + e;
    float v = 0.f;
    if (k < K) v = TA ? A[(long long)k * lda + gm] : A[(long long)gm * lda + k];
    a[e] = (_Float16)v;
  }
  return a;
}

template <int TB>
__device__ inline v16h frag_b_tail(const float* __restrict__ B, int ldb, int gn,
                                   int kk, int half, int K) {
  v16h b;
#pragma unroll
  for (int e = 0; e < 16; ++e) {
    int k = kk + 16 * half + e;
    float v = 0.f;
    if (k < K) v = TB ? B[(long long)gn * ldb + k] : B[(long long)k * ldb + gn];
    b[e] = (_Float16)v;
  }
  return b;
}

// ---------------------------------------------------------------------------
// Batched GEMM, compile-time layouts. M, N must be multiples of 16 (true for
// every call site). K tail handled by one guarded chunk. One wave per tile.
// ---------------------------------------------------------------------------
template <int TA, int TB>
__global__ void gemm_wmma_t(const float* __restrict__ A, long long sA, int lda,
                            const float* __restrict__ B, long long sB, int ldb,
                            const float* __restrict__ bias,
                            float* __restrict__ C, long long sC, int ldc, int K) {
  const float* Ab = A + (long long)blockIdx.z * sA;
  const float* Bb = B + (long long)blockIdx.z * sB;
  float* Cb = C + (long long)blockIdx.z * sC;
  const int tm = blockIdx.x * 16, tn = blockIdx.y * 16;
  const int lane = threadIdx.x & 31;
  const int half = lane >> 4;
  const int ml = lane & 15;
  const int gm = tm + ml, gn = tn + ml;
  const float* Arow = Ab + (long long)gm * lda;  // used when TA==0
  const float* Brow = Bb + (long long)gn * ldb;  // used when TB==1
  v8f acc = {};
  const int Kfull = K & ~31;
  for (int kk = 0; kk < Kfull; kk += 32) {
    v16h a = TA ? frag_a_kstrided(Ab, lda, gm, kk, half) : frag_a_rowmajor(Arow, kk, half);
    v16h b = TB ? frag_b_rowmajor(Brow, kk, half) : frag_b_kstrided(Bb, ldb, gn, kk, half);
    acc = __builtin_amdgcn_wmma_f32_16x16x32_f16(false, a, false, b, (short)0, acc,
                                                 false, false);
  }
  if (Kfull < K) {
    v16h a = frag_a_tail<TA>(Ab, lda, gm, Kfull, half, K);
    v16h b = frag_b_tail<TB>(Bb, ldb, gn, Kfull, half, K);
    acc = __builtin_amdgcn_wmma_f32_16x16x32_f16(false, a, false, b, (short)0, acc,
                                                 false, false);
  }
  float bz = bias ? bias[gn] : 0.f;
#pragma unroll
  for (int r = 0; r < 8; ++r)
    Cb[(long long)(tm + r + 8 * half) * ldc + gn] = acc[r] + bz;
}

// ---------------------------------------------------------------------------
// Persistent LSTM: one workgroup per direction (gridDim.x == 2), batch fixed
// at 8 (left+right or mv_p+mv_h stacked), S = 256, 1024 threads.
// Matmul phase stores raw WMMA accumulators; the pre[] addend is applied in
// the gate phase with fully coalesced unpredicated loads.
// ---------------------------------------------------------------------------
__global__ void lstm_wmma(const float* __restrict__ preF, const float* __restrict__ preB,
                          const float* __restrict__ WhhF, const float* __restrict__ WhhB,
                          float* __restrict__ hsF, float* __restrict__ hsB,
                          float* __restrict__ hTF, float* __restrict__ hTB) {
  const int dir = blockIdx.x;
  const float* pre = dir ? preB : preF;
  const float* Whh = dir ? WhhB : WhhF;
  float* hs = dir ? hsB : hsF;
  float* hT = dir ? hTB : hTF;

  __shared__ __attribute__((aligned(16))) _Float16 hH[16 * Hh];  // 8 KB
  __shared__ float hF[8 * Hh];    // 8 KB
  __shared__ float cF[8 * Hh];    // 8 KB
  __shared__ float gAll[8 * G4];  // 32 KB

  const int tid = threadIdx.x;
#pragma unroll
  for (int it = 0; it < 4; ++it) hH[tid + it * 1024] = (_Float16)0.f;
#pragma unroll
  for (int it = 0; it < 2; ++it) { hF[tid + it * 1024] = 0.f; cF[tid + it * 1024] = 0.f; }
  __syncthreads();

  const int lane = tid & 31, wave = tid >> 5;
  const int half = lane >> 4, ml = lane & 15;

  for (int step = 0; step < SQ; ++step) {
    const int t = dir ? (SQ - 1 - step) : step;
    int tpre = dir ? (t - 1) : (t + 1);
    tpre = tpre < 0 ? 0 : (tpre > SQ - 1 ? SQ - 1 : tpre);
    // one prefetch per step: waves 0..7 pull next step's pre row into caches
    if (wave < 8)
      __builtin_prefetch(pre + ((long long)wave * SQ + tpre) * G4 + lane * 32, 0, 0);
    // ---- matmul: g = h @ Whh^T, 64 N-tiles over 32 waves (2 each) ----
#pragma unroll
    for (int half_tile = 0; half_tile < 2; ++half_tile) {
      const int tn = (wave + half_tile * 32) * 16;
      const float* wrow = Whh + (long long)(tn + ml) * Hh;
      v8f acc = {};
#pragma unroll 2
      for (int kk = 0; kk < Hh; kk += 32) {
        const v8h* A0 = (const v8h*)&hH[ml * Hh + kk + 16 * half];
        const v8h* A1 = (const v8h*)&hH[ml * Hh + kk + 16 * half + 8];
        v16h a = cat8(*A0, *A1);
        const float4* q = (const float4*)(wrow + kk + 16 * half);
        v16h b;
#pragma unroll
        for (int e = 0; e < 4; ++e) {
          float4 x = q[e];
          b[4 * e] = (_Float16)x.x; b[4 * e + 1] = (_Float16)x.y;
          b[4 * e + 2] = (_Float16)x.z; b[4 * e + 3] = (_Float16)x.w;
        }
        acc = __builtin_amdgcn_wmma_f32_16x16x32_f16(false, a, false, b, (short)0,
                                                     acc, false, false);
      }
      if (half == 0) {  // rows 0..7 live in lanes 0-15; batch is exactly 8
#pragma unroll
        for (int r = 0; r < 8; ++r) gAll[r * G4 + tn + ml] = acc[r];
      }
    }
    __syncthreads();
    // ---- gates (i,f,g,o), pre added here with coalesced loads ----
#pragma unroll
    for (int it = 0; it < 2; ++it) {
      const int j = tid + it * 1024;
      const int m = j >> 8, u = j & 255;
      const float* prow = pre + ((long long)m * SQ + t) * G4;
      float gi = gAll[m * G4 + u] + prow[u];
      float gf = gAll[m * G4 + 256 + u] + prow[256 + u];
      float gg = gAll[m * G4 + 512 + u] + prow[512 + u];
      float go = gAll[m * G4 + 768 + u] + prow[768 + u];
      float c = sigm(gf) * cF[j] + sigm(gi) * tanhf(gg);
      float h = sigm(go) * tanhf(c);
      cF[j] = c;
      hF[j] = h;
      hH[m * Hh + u] = (_Float16)h;
      hs[((long long)m * SQ + t) * Hh + u] = h;
    }
    __syncthreads();
  }
#pragma unroll
  for (int it = 0; it < 2; ++it) hT[tid + it * 1024] = hF[tid + it * 1024];
}

// ---------------------------------------------------------------------------
// Fused mp_match_pairwise + max over V2's axis. One wave per (b, l, 16 rows
// of V1); both operands are K-contiguous -> float4 loads, scaled by w_l.
// ---------------------------------------------------------------------------
__global__ void mpmax_wmma(const float* __restrict__ V1, const float* __restrict__ V2,
                           const float* __restrict__ w,
                           const float* __restrict__ n1, const float* __restrict__ n2,
                           float* __restrict__ out, int outStride, int outOff, int S) {
  const int b = blockIdx.z, l = blockIdx.y, tp = blockIdx.x * 16;
  const int lane = threadIdx.x & 31, half = lane >> 4, ml = lane & 15;
  const float* v1row = V1 + ((long long)b * S + tp + ml) * Hh;
  const float* v2b = V2 + (long long)b * S * Hh;
  const float* wl = w + l * Hh;
  const float* n1b = n1 + ((long long)b * LL + l) * S;
  const float* n2b = n2 + ((long long)b * LL + l) * S;
  float rm[8];
#pragma unroll
  for (int r = 0; r < 8; ++r) rm[r] = -1e30f;
  for (int tq = 0; tq < S; tq += 16) {
    const float* v2row = v2b + (long long)(tq + ml) * Hh;
    v8f acc = {};
    for (int kk = 0; kk < Hh; kk += 32) {
      v16h a, bf;
      {  // A: one 16-elem run at kk+16*half (two 8-elem halves)
        const float4* p0 = (const float4*)(v1row + kk + 16 * half);
        const float4* w0 = (const float4*)(wl + kk + 16 * half);
        const float4* p1 = (const float4*)(v1row + kk + 16 * half + 8);
        const float4* w1 = (const float4*)(wl + kk + 16 * half + 8);
#pragma unroll
        for (int e = 0; e < 2; ++e) {
          float4 x = p0[e], y = w0[e];
          a[4 * e] = (_Float16)(x.x * y.x); a[4 * e + 1] = (_Float16)(x.y * y.y);
          a[4 * e + 2] = (_Float16)(x.z * y.z); a[4 * e + 3] = (_Float16)(x.w * y.w);
          float4 u = p1[e], v = w1[e];
          a[8 + 4 * e] = (_Float16)(u.x * v.x); a[8 + 4 * e + 1] = (_Float16)(u.y * v.y);
          a[8 + 4 * e + 2] = (_Float16)(u.z * v.z); a[8 + 4 * e + 3] = (_Float16)(u.w * v.w);
        }
      }
      {  // B: one 16-elem run at kk+16*half
        const float4* p = (const float4*)(v2row + kk + 16 * half);
        const float4* q = (const float4*)(wl + kk + 16 * half);
#pragma unroll
        for (int e = 0; e < 4; ++e) {
          float4 x = p[e], y = q[e];
          bf[4 * e] = (_Float16)(x.x * y.x); bf[4 * e + 1] = (_Float16)(x.y * y.y);
          bf[4 * e + 2] = (_Float16)(x.z * y.z); bf[4 * e + 3] = (_Float16)(x.w * y.w);
        }
      }
      acc = __builtin_amdgcn_wmma_f32_16x16x32_f16(false, a, false, bf, (short)0,
                                                   acc, false, false);
    }
#pragma unroll
    for (int r = 0; r < 8; ++r) {
      float d = n1b[tp + r + 8 * half] * n2b[tq + ml];
      float cv = acc[r] / (d > EPSV ? d : EPSV);
      rm[r] = fmaxf(rm[r], cv);
    }
  }
#pragma unroll
  for (int off = 8; off >= 1; off >>= 1) {
#pragma unroll
    for (int r = 0; r < 8; ++r) rm[r] = fmaxf(rm[r], __shfl_xor(rm[r], off, 32));
  }
  if (ml == 0) {
#pragma unroll
    for (int r = 0; r < 8; ++r)
      out[((long long)b * S + tp + r + 8 * half) * outStride + outOff + l] = rm[r];
  }
}

// ------------------------ small VALU helper kernels ------------------------
__global__ void rownorm_kernel(const float* __restrict__ V, float* __restrict__ out,
                               int rows) {
  int idx = blockIdx.x * blockDim.x + threadIdx.x;
  if (idx >= rows) return;
  const float* v = V + (long long)idx * Hh;
  float acc = 0.f;
  for (int h = 0; h < Hh; ++h) acc += v[h] * v[h];
  out[idx] = sqrtf(acc);
}

__global__ void wnorm_kernel(const float* __restrict__ V, const float* __restrict__ w,
                             float* __restrict__ out, int S) {
  int idx = blockIdx.x * blockDim.x + threadIdx.x;  // (b, l, s)
  if (idx >= BB * LL * S) return;
  int s = idx % S, l = (idx / S) % LL, b = idx / (S * LL);
  const float* v = V + ((long long)b * S + s) * Hh;
  const float* wl = w + l * Hh;
  float acc = 0.f;
  for (int h = 0; h < Hh; ++h) { float x = wl[h] * v[h]; acc += x * x; }
  out[idx] = sqrtf(acc);
}

__global__ void attnorm_kernel(float* __restrict__ att, const float* __restrict__ np,
                               const float* __restrict__ nh, int S) {
  int idx = blockIdx.x * blockDim.x + threadIdx.x;
  if (idx >= BB * S * S) return;
  int b = idx / (S * S), r = idx % (S * S), p = r / S, q = r % S;
  float d = np[b * S + p] * nh[b * S + q];
  att[idx] = att[idx] / (d > EPSV ? d : EPSV);
}

__global__ void att_reduce(const float* __restrict__ att, int sI, int sJ, int S,
                           float* outMax, int mStride, int mOff,
                           float* outMean, int eStride, int eOff, float* outSum) {
  int b = blockIdx.y, i = blockIdx.x, tid = threadIdx.x;  // blockDim == S == 256
  float v = att[(long long)b * S * S + (long long)i * sI + (long long)tid * sJ];
  __shared__ float sm[256], ss[256];
  sm[tid] = v; ss[tid] = v;
  __syncthreads();
  for (int off = 128; off; off >>= 1) {
    if (tid < off) { sm[tid] = fmaxf(sm[tid], sm[tid + off]); ss[tid] += ss[tid + off]; }
    __syncthreads();
  }
  if (tid == 0) {
    long long o = (long long)b * S + i;
    if (outMax) outMax[o * mStride + mOff] = sm[0];
    if (outMean) outMean[o * eStride + eOff] = ss[0] / (float)S;
    if (outSum) outSum[o] = ss[0];
  }
}

__global__ void divrow_kernel(float* __restrict__ X, const float* __restrict__ d, int S) {
  int idx = blockIdx.x * blockDim.x + threadIdx.x;  // (b, i, h)
  if (idx >= BB * S * Hh) return;
  int b = idx / (S * Hh), i = (idx / Hh) % S;
  float dd = d[b * S + i];
  X[idx] = X[idx] / (dd > EPSV ? dd : EPSV);
}

__global__ void attmax_kernel(const float* __restrict__ att, int sI, int sJ,
                              const float* __restrict__ V, float* __restrict__ out,
                              int S) {
  int bi = blockIdx.x, b = bi / S, i = bi % S, tid = threadIdx.x;  // tid == h
  __shared__ float row[256];
  row[tid] = att[(long long)b * S * S + (long long)i * sI + (long long)tid * sJ];
  __syncthreads();
  float m = -1e30f;
  const float* Vb = V + (long long)b * S * Hh;
  for (int j = 0; j < S; ++j) m = fmaxf(m, row[j] * Vb[j * Hh + tid]);
  out[((long long)b * S + i) * Hh + tid] = m;
}

__global__ void mpmatch_kernel(const float* __restrict__ V1, const float* __restrict__ V2,
                               const float* __restrict__ w, float* __restrict__ out,
                               int outStride, int outOff, int S) {
  int idx = blockIdx.x * blockDim.x + threadIdx.x;  // (b, s, l)
  if (idx >= BB * S * LL) return;
  int l = idx % LL, s = (idx / LL) % S, b = idx / (LL * S);
  const float* a = V1 + ((long long)b * S + s) * Hh;
  const float* c = V2 + ((long long)b * S + s) * Hh;
  const float* wl = w + l * Hh;
  float num = 0.f, q1 = 0.f, q2 = 0.f;
  for (int h = 0; h < Hh; ++h) {
    float w2 = wl[h] * wl[h];
    num += w2 * a[h] * c[h];
    q1 += w2 * a[h] * a[h];
    q2 += w2 * c[h] * c[h];
  }
  out[((long long)b * S + s) * outStride + outOff + l] =
      num / fmaxf(sqrtf(q1) * sqrtf(q2), EPSV);
}

__global__ void seqmean_kernel(const float* __restrict__ X, float* __restrict__ out,
                               int S, int D) {
  int idx = blockIdx.x * blockDim.x + threadIdx.x;
  if (idx >= BB * D) return;
  int d = idx % D, b = idx / D;
  float s = 0.f;
  for (int t = 0; t < S; ++t) s += X[((long long)b * S + t) * D + d];
  out[idx] = s / (float)S;
}

__global__ void fc_kernel(const float* __restrict__ hTF, const float* __restrict__ hTB,
                          const float* __restrict__ meanL, const float* __restrict__ meanR,
                          const float* __restrict__ W1, const float* __restrict__ b1,
                          const float* __restrict__ W2, const float* __restrict__ b2,
                          float* __restrict__ out) {
  const int b = blockIdx.x, tid = threadIdx.x;  // blockDim == 512
  const int XD = 4 * Hh + 2 + 2 * DD;           // 1626
  __shared__ float x[4 * Hh + 2 + 2 * DD];
  __shared__ float y1[512];
  for (int j = tid; j < XD; j += blockDim.x) {
    float v;
    if (j < 256)       v = hTF[b * Hh + j];               // ap_f
    else if (j < 512)  v = hTB[b * Hh + (j - 256)];       // ap_b
    else if (j < 768)  v = hTF[(4 + b) * Hh + (j - 512)]; // ah_f
    else if (j < 1024) v = hTB[(4 + b) * Hh + (j - 768)]; // ah_b
    else if (j < 1026) v = 0.5f;                          // lengths (256/512)
    else if (j < 1326) v = meanL[b * DD + (j - 1026)];
    else               v = meanR[b * DD + (j - 1326)];
    x[j] = v;
  }
  __syncthreads();
  {
    float acc = b1[tid];
    const float* wr = W1 + (long long)tid * XD;
    for (int k = 0; k < XD; ++k) acc += wr[k] * x[k];
    y1[tid] = tanhf(acc);
  }
  __syncthreads();
  if (tid < NCLSV) {
    float acc = b2[tid];
    const float* wr = W2 + tid * 512;
    for (int k = 0; k < 512; ++k) acc += wr[k] * y1[k];
    out[b * NCLSV + tid] = acc;
  }
}

// ---------------------------------------------------------------------------
extern "C" void kernel_launch(void* const* d_in, const int* in_sizes, int n_in,
                              void* d_out, int out_size, void* d_ws, size_t ws_size,
                              hipStream_t stream) {
  (void)in_sizes; (void)n_in; (void)out_size; (void)ws_size;
  const float* left      = (const float*)d_in[0];
  const float* right     = (const float*)d_in[1];
  const float* ctx_Wih_f = (const float*)d_in[2];
  const float* ctx_Whh_f = (const float*)d_in[3];
  const float* ctx_b_f   = (const float*)d_in[4];
  const float* ctx_Wih_b = (const float*)d_in[5];
  const float* ctx_Whh_b = (const float*)d_in[6];
  const float* ctx_b_b   = (const float*)d_in[7];
  const float* agg_Wih_f = (const float*)d_in[8];
  const float* agg_Whh_f = (const float*)d_in[9];
  const float* agg_b_f   = (const float*)d_in[10];
  const float* agg_Wih_b = (const float*)d_in[11];
  const float* agg_Whh_b = (const float*)d_in[12];
  const float* agg_b_b   = (const float*)d_in[13];
  const float* w3 = (const float*)d_in[14];
  const float* w4 = (const float*)d_in[15];
  const float* w5 = (const float*)d_in[16];
  const float* w6 = (const float*)d_in[17];
  const float* w7 = (const float*)d_in[18];
  const float* w8 = (const float*)d_in[19];
  const float* fc1_W = (const float*)d_in[20];
  const float* fc1_b = (const float*)d_in[21];
  const float* fc2_W = (const float*)d_in[22];
  const float* fc2_b = (const float*)d_in[23];

  float* ws = (float*)d_ws;
  size_t off = 0;
  auto alloc = [&](size_t n) { float* p = ws + off; off += (n + 63) & ~(size_t)63; return p; };

  float* preCtxF = alloc(8ull * SQ * G4);
  float* preCtxB = alloc(8ull * SQ * G4);
  float* hsF     = alloc(8ull * SQ * Hh);
  float* hsB     = alloc(8ull * SQ * Hh);
  float* hTctxF  = alloc(8 * Hh);
  float* hTctxB  = alloc(8 * Hh);
  float* nF      = alloc(8 * SQ);
  float* nB      = alloc(8 * SQ);
  float* attF    = alloc((size_t)BB * SQ * SQ);
  float* attB    = alloc((size_t)BB * SQ * SQ);
  float* rowsumF = alloc(BB * SQ);
  float* colsumF = alloc(BB * SQ);
  float* rowsumB = alloc(BB * SQ);
  float* colsumB = alloc(BB * SQ);
  float* nw3p = alloc(BB * LL * SQ);
  float* nw3h = alloc(BB * LL * SQ);
  float* nw4p = alloc(BB * LL * SQ);
  float* nw4h = alloc(BB * LL * SQ);
  float* meanHf = alloc((size_t)BB * SQ * Hh);
  float* meanPf = alloc((size_t)BB * SQ * Hh);
  float* meanHb = alloc((size_t)BB * SQ * Hh);
  float* meanPb = alloc((size_t)BB * SQ * Hh);
  float* maxHf  = alloc((size_t)BB * SQ * Hh);
  float* maxPf  = alloc((size_t)BB * SQ * Hh);
  float* maxHb  = alloc((size_t)BB * SQ * Hh);
  float* maxPb  = alloc((size_t)BB * SQ * Hh);
  float* mv      = alloc(8ull * SQ * AGGIN);   // rows 0..1023 = mv_p, 1024..2047 = mv_h
  float* preAggF = alloc(8ull * SQ * G4);
  float* preAggB = alloc(8ull * SQ * G4);
  float* hsAggF  = alloc(8ull * SQ * Hh);
  float* hsAggB  = alloc(8ull * SQ * Hh);
  float* hTaggF  = alloc(8 * Hh);
  float* hTaggB  = alloc(8 * Hh);
  float* meanL   = alloc(BB * DD);
  float* meanR   = alloc(BB * DD);

  const dim3 w32(32);
  const long long sSH = (long long)SQ * Hh;
  const long long sSS = (long long)SQ * SQ;

  // --- context input projections (left+right stacked as batch 8) ---
  gemm_wmma_t<0, 1><<<dim3(64, 64, 1), w32, 0, stream>>>(left, 0, DD, ctx_Wih_f, 0, DD,
      ctx_b_f, preCtxF, 0, G4, DD);
  gemm_wmma_t<0, 1><<<dim3(64, 64, 1), w32, 0, stream>>>(right, 0, DD, ctx_Wih_f, 0, DD,
      ctx_b_f, preCtxF + 4ull * SQ * G4, 0, G4, DD);
  gemm_wmma_t<0, 1><<<dim3(64, 64, 1), w32, 0, stream>>>(left, 0, DD, ctx_Wih_b, 0, DD,
      ctx_b_b, preCtxB, 0, G4, DD);
  gemm_wmma_t<0, 1><<<dim3(64, 64, 1), w32, 0, stream>>>(right, 0, DD, ctx_Wih_b, 0, DD,
      ctx_b_b, preCtxB + 4ull * SQ * G4, 0, G4, DD);

  // --- context BiLSTM recurrence (2 workgroups: fw, bw) ---
  lstm_wmma<<<2, 1024, 0, stream>>>(preCtxF, preCtxB, ctx_Whh_f, ctx_Whh_b,
                                    hsF, hsB, hTctxF, hTctxB);
  const float* p_fw = hsF;
  const float* h_fw = hsF + 4ull * SQ * Hh;
  const float* p_bw = hsB;
  const float* h_bw = hsB + 4ull * SQ * Hh;

  // --- row norms, cosine matrices ---
  rownorm_kernel<<<(8 * SQ + 255) / 256, 256, 0, stream>>>(hsF, nF, 8 * SQ);
  rownorm_kernel<<<(8 * SQ + 255) / 256, 256, 0, stream>>>(hsB, nB, 8 * SQ);
  gemm_wmma_t<0, 1><<<dim3(16, 16, BB), w32, 0, stream>>>(p_fw, sSH, Hh, h_fw, sSH, Hh,
      nullptr, attF, sSS, SQ, Hh);
  gemm_wmma_t<0, 1><<<dim3(16, 16, BB), w32, 0, stream>>>(p_bw, sSH, Hh, h_bw, sSH, Hh,
      nullptr, attB, sSS, SQ, Hh);
  attnorm_kernel<<<(BB * SQ * SQ + 255) / 256, 256, 0, stream>>>(attF, nF, nF + 4 * SQ, SQ);
  attnorm_kernel<<<(BB * SQ * SQ + 255) / 256, 256, 0, stream>>>(attB, nB, nB + 4 * SQ, SQ);

  // --- att_fw max/mean -> mv cols 0,1; sums for attentive normalization ---
  att_reduce<<<dim3(SQ, BB), 256, 0, stream>>>(attF, SQ, 1, SQ, mv, AGGIN, 0,
      mv, AGGIN, 1, rowsumF);
  att_reduce<<<dim3(SQ, BB), 256, 0, stream>>>(attF, 1, SQ, SQ, mv + 4ull * SQ * AGGIN,
      AGGIN, 0, mv + 4ull * SQ * AGGIN, AGGIN, 1, colsumF);
  att_reduce<<<dim3(SQ, BB), 256, 0, stream>>>(attB, SQ, 1, SQ, nullptr, 0, 0,
      nullptr, 0, 0, rowsumB);
  att_reduce<<<dim3(SQ, BB), 256, 0, stream>>>(attB, 1, SQ, SQ, nullptr, 0, 0,
      nullptr, 0, 0, colsumB);

  // --- maxpool (pairwise) match, cols 2..21 ---
  int nw = BB * LL * SQ;
  wnorm_kernel<<<(nw + 255) / 256, 256, 0, stream>>>(p_fw, w3, nw3p, SQ);
  wnorm_kernel<<<(nw + 255) / 256, 256, 0, stream>>>(h_fw, w3, nw3h, SQ);
  wnorm_kernel<<<(nw + 255) / 256, 256, 0, stream>>>(p_bw, w4, nw4p, SQ);
  wnorm_kernel<<<(nw + 255) / 256, 256, 0, stream>>>(h_bw, w4, nw4h, SQ);
  float* mvP = mv;
  float* mvH = mv + 4ull * SQ * AGGIN;
  mpmax_wmma<<<dim3(16, LL, BB), w32, 0, stream>>>(p_fw, h_fw, w3, nw3p, nw3h, mvP, AGGIN, 2, SQ);
  mpmax_wmma<<<dim3(16, LL, BB), w32, 0, stream>>>(p_bw, h_bw, w4, nw4p, nw4h, mvP, AGGIN, 12, SQ);
  mpmax_wmma<<<dim3(16, LL, BB), w32, 0, stream>>>(h_fw, p_fw, w3, nw3h, nw3p, mvH, AGGIN, 2, SQ);
  mpmax_wmma<<<dim3(16, LL, BB), w32, 0, stream>>>(h_bw, p_bw, w4, nw4h, nw4p, mvH, AGGIN, 12, SQ);

  // --- attentive means (GEMM + divide) ---
  gemm_wmma_t<0, 0><<<dim3(16, 16, BB), w32, 0, stream>>>(attF, sSS, SQ, h_fw, sSH, Hh,
      nullptr, meanHf, sSH, Hh, SQ);
  gemm_wmma_t<1, 0><<<dim3(16, 16, BB), w32, 0, stream>>>(attF, sSS, SQ, p_fw, sSH, Hh,
      nullptr, meanPf, sSH, Hh, SQ);
  gemm_wmma_t<0, 0><<<dim3(16, 16, BB), w32, 0, stream>>>(attB, sSS, SQ, h_bw, sSH, Hh,
      nullptr, meanHb, sSH, Hh, SQ);
  gemm_wmma_t<1, 0><<<dim3(16, 16, BB), w32, 0, stream>>>(attB, sSS, SQ, p_bw, sSH, Hh,
      nullptr, meanPb, sSH, Hh, SQ);
  int ndiv = BB * SQ * Hh;
  divrow_kernel<<<(ndiv + 255) / 256, 256, 0, stream>>>(meanHf, rowsumF, SQ);
  divrow_kernel<<<(ndiv + 255) / 256, 256, 0, stream>>>(meanPf, colsumF, SQ);
  divrow_kernel<<<(ndiv + 255) / 256, 256, 0, stream>>>(meanHb, rowsumB, SQ);
  divrow_kernel<<<(ndiv + 255) / 256, 256, 0, stream>>>(meanPb, colsumB, SQ);

  // --- max-attentive vectors ---
  attmax_kernel<<<BB * SQ, 256, 0, stream>>>(attF, SQ, 1, h_fw, maxHf, SQ);
  attmax_kernel<<<BB * SQ, 256, 0, stream>>>(attF, 1, SQ, p_fw, maxPf, SQ);
  attmax_kernel<<<BB * SQ, 256, 0, stream>>>(attB, SQ, 1, h_bw, maxHb, SQ);
  attmax_kernel<<<BB * SQ, 256, 0, stream>>>(attB, 1, SQ, p_bw, maxPb, SQ);

  // --- attentive / max-attentive per-position matching, cols 22..61 ---
  int nmp = BB * SQ * LL;
  mpmatch_kernel<<<(nmp + 255) / 256, 256, 0, stream>>>(p_fw, meanHf, w5, mvP, AGGIN, 22, SQ);
  mpmatch_kernel<<<(nmp + 255) / 256, 256, 0, stream>>>(p_bw, meanHb, w6, mvP, AGGIN, 32, SQ);
  mpmatch_kernel<<<(nmp + 255) / 256, 256, 0, stream>>>(p_fw, maxHf, w7, mvP, AGGIN, 42, SQ);
  mpmatch_kernel<<<(nmp + 255) / 256, 256, 0, stream>>>(p_bw, maxHb, w8, mvP, AGGIN, 52, SQ);
  mpmatch_kernel<<<(nmp + 255) / 256, 256, 0, stream>>>(h_fw, meanPf, w5, mvH, AGGIN, 22, SQ);
  mpmatch_kernel<<<(nmp + 255) / 256, 256, 0, stream>>>(h_bw, meanPb, w6, mvH, AGGIN, 32, SQ);
  mpmatch_kernel<<<(nmp + 255) / 256, 256, 0, stream>>>(h_fw, maxPf, w7, mvH, AGGIN, 42, SQ);
  mpmatch_kernel<<<(nmp + 255) / 256, 256, 0, stream>>>(h_bw, maxPb, w8, mvH, AGGIN, 52, SQ);

  // --- aggregation input projections (mv_p + mv_h stacked, M = 2048) ---
  gemm_wmma_t<0, 1><<<dim3(128, 64, 1), w32, 0, stream>>>(mv, 0, AGGIN, agg_Wih_f, 0, AGGIN,
      agg_b_f, preAggF, 0, G4, AGGIN);
  gemm_wmma_t<0, 1><<<dim3(128, 64, 1), w32, 0, stream>>>(mv, 0, AGGIN, agg_Wih_b, 0, AGGIN,
      agg_b_b, preAggB, 0, G4, AGGIN);

  // --- aggregation BiLSTM (only final hiddens consumed) ---
  lstm_wmma<<<2, 1024, 0, stream>>>(preAggF, preAggB, agg_Whh_f, agg_Whh_b,
                                    hsAggF, hsAggB, hTaggF, hTaggB);

  // --- sequence means + final FC ---
  seqmean_kernel<<<(BB * DD + 255) / 256, 256, 0, stream>>>(left, meanL, SQ, DD);
  seqmean_kernel<<<(BB * DD + 255) / 256, 256, 0, stream>>>(right, meanR, SQ, DD);
  fc_kernel<<<BB, 512, 0, stream>>>(hTaggF, hTaggB, meanL, meanR, fc1_W, fc1_b,
                                    fc2_W, fc2_b, (float*)d_out);
}
